// InsZeroShotClassifier_25778393711040
// MI455X (gfx1250) — matrix operand order repaired and checked by reading, not compiled, not run on hardware
//
#include <hip/hip_runtime.h>
#include <hip/hip_bf16.h>
#include <cstdint>
#include <cstddef>

// ---------------- problem constants ----------------
#define B_ROWS   1000
#define DIN_     1024
#define D_       512
#define NCLS_    1204      // C+1
#define NC_      1203
#define LSEQ_    77
#define NINS_    4
#define NEX_     5
#define KSEL_    32
#define KTOP_    5
#define NSEQ_    160       // KSEL*KTOP
#define HEADS_   8
#define DH_      64
#define MLP_     2048
#define NL_      4
#define TEMP_F   50.0f

typedef __attribute__((ext_vector_type(16))) _Float16 v16h;
typedef __attribute__((ext_vector_type(8)))  _Float16 v8h;
typedef __attribute__((ext_vector_type(8)))  float    v8f;

// =====================================================================
// WMMA GEMM:  C[M,N] = epilogue(A[M,K] @ W[K,N])
//   A  : f16 row-major [M,K]
//   WT : f16 row-major [N,K]   (i.e., W transposed, pre-converted)
//   epilogue: (+bias[N]) -> (QuickGELU) -> (+resid[M,N] f32); out f32 or f16
// One wave computes a 16x64 strip (4 accumulators, A fragment reused 4x).
// All loads of a k-step are hoisted ahead of the 4 WMMAs so the scheduler
// can issue one load clause and use partial s_wait_loadcnt before each WMMA.
// Edge tiles: clamped addresses only — garbage lands in masked C rows/cols.
// =====================================================================
template<bool BIAS, bool RESID, bool GELU, bool OUT16>
__global__ __launch_bounds__(256)
void gemm_wmma_k(const _Float16* __restrict__ A, const _Float16* __restrict__ WT,
                 const float* __restrict__ bias, const float* __restrict__ resid,
                 void* __restrict__ Cv, int M, int N, int K) {
    const int lane    = threadIdx.x & 31;
    const int wave    = threadIdx.x >> 5;
    const int stripsN = (N + 63) >> 6;
    const int tilesM  = (M + 15) >> 4;
    const int strip   = blockIdx.x * 8 + wave;
    if (strip >= tilesM * stripsN) return;
    const int tm  = strip / stripsN;
    const int tn0 = (strip % stripsN) << 2;        // first 16-wide tile in strip
    const int r   = lane & 15;
    const int kh  = lane >> 4;                     // lane half (0/1)
    const int row  = tm * 16 + r;
    const int rowC = (row < M) ? row : (M - 1);
    const _Float16* __restrict__ Ar = A + (size_t)rowC * K;

    const _Float16* __restrict__ Bc[4];
    #pragma unroll
    for (int j = 0; j < 4; ++j) {
        const int col  = (tn0 + j) * 16 + r;
        const int colC = (col < N) ? col : (N - 1);
        Bc[j] = WT + (size_t)colC * K;
    }

    v8f acc[4] = {};
    for (int k0 = 0; k0 < K; k0 += 32) {
        // ---- issue ALL loads of this k-step up front (one clause) ----
        const v8h a0 = *(const v8h*)(Ar + k0 + (kh << 3));
        const v8h a1 = *(const v8h*)(Ar + k0 + 16 + (kh << 3));
        v8h b0[4], b1[4];
        #pragma unroll
        for (int j = 0; j < 4; ++j) {
            b0[j] = *(const v8h*)(Bc[j] + k0 + (kh << 3));
            b1[j] = *(const v8h*)(Bc[j] + k0 + 16 + (kh << 3));
        }
        __builtin_prefetch(Ar + k0 + 256, 0, 1);
        // ---- 4 WMMAs; scheduler can use partial waits per accumulator ----
        const v16h a = __builtin_shufflevector(a0, a1,
            0, 1, 2, 3, 4, 5, 6, 7, 8, 9, 10, 11, 12, 13, 14, 15);
        #pragma unroll
        for (int j = 0; j < 4; ++j) {
            const v16h b = __builtin_shufflevector(b0[j], b1[j],
                0, 1, 2, 3, 4, 5, 6, 7, 8, 9, 10, 11, 12, 13, 14, 15);
            acc[j] = __builtin_amdgcn_wmma_f32_16x16x32_f16(
                false, a, false, b, (short)0, acc[j], false, false);
        }
    }

    float*    Cf = (float*)Cv;
    _Float16* Ch = (_Float16*)Cv;
    #pragma unroll
    for (int j = 0; j < 4; ++j) {
        const int ocol = (tn0 + j) * 16 + r;
        if (ocol < N) {
            #pragma unroll
            for (int v = 0; v < 8; ++v) {
                const int orow = tm * 16 + v + (kh << 3);
                if (orow < M) {
                    float o = acc[j][v];
                    if (BIAS)  o += bias[ocol];
                    if (GELU)  o = o / (1.f + __expf(-1.702f * o));   // QuickGELU
                    if (RESID) o += resid[(size_t)orow * N + ocol];
                    if (OUT16) Ch[(size_t)orow * N + ocol] = (_Float16)o;
                    else       Cf[(size_t)orow * N + ocol] = o;
                }
            }
        }
    }
}

// =====================================================================
// Weight pre-pass: W f32 [K,N]  ->  WT f16 [N,K]   (LDS-tiled transpose)
// =====================================================================
__global__ __launch_bounds__(256)
void transpose_cvt_k(const float* __restrict__ W, _Float16* __restrict__ WT,
                     int K, int N) {
    __shared__ float tile[32][33];
    const int kb = blockIdx.x * 32, nb = blockIdx.y * 32;
    const int tx = threadIdx.x & 31, ty = threadIdx.x >> 5;   // 32 x 8
    for (int i = ty; i < 32; i += 8) {
        const int k = kb + i, n = nb + tx;
        tile[i][tx] = (k < K && n < N) ? W[(size_t)k * N + n] : 0.f;
    }
    __syncthreads();
    for (int i = ty; i < 32; i += 8) {
        const int n = nb + i, k = kb + tx;
        if (n < N && k < K) WT[(size_t)n * K + k] = (_Float16)tile[tx][i];
    }
}

__global__ void cvt_f16_k(const float* __restrict__ src, _Float16* __restrict__ dst, int n) {
    const int i = blockIdx.x * blockDim.x + threadIdx.x;
    if (i < n) dst[i] = (_Float16)src[i];
}

// =====================================================================
// LayerNorm over last dim (D=512). One block per row. f32 in, f16 out.
// =====================================================================
__global__ __launch_bounds__(128)
void layernorm_k(const float* __restrict__ X, const float* __restrict__ g,
                 const float* __restrict__ b, _Float16* __restrict__ Y, int R) {
    __shared__ float s1[128], s2[128];
    const int row = blockIdx.x;
    if (row >= R) return;
    const float* x = X + (size_t)row * D_;
    float a = 0.f, q = 0.f;
    for (int i = threadIdx.x; i < D_; i += 128) { float v = x[i]; a += v; q += v * v; }
    s1[threadIdx.x] = a; s2[threadIdx.x] = q; __syncthreads();
    for (int s = 64; s > 0; s >>= 1) {
        if (threadIdx.x < s) { s1[threadIdx.x] += s1[threadIdx.x + s];
                               s2[threadIdx.x] += s2[threadIdx.x + s]; }
        __syncthreads();
    }
    const float mean = s1[0] * (1.f / D_);
    const float var  = s2[0] * (1.f / D_) - mean * mean;
    const float inv  = rsqrtf(var + 1e-5f);
    for (int i = threadIdx.x; i < D_; i += 128)
        Y[(size_t)row * D_ + i] = (_Float16)((x[i] - mean) * inv * g[i] + b[i]);
}

// =====================================================================
// Row L2-normalize in place (f32):  x *= numer / ||x||
// =====================================================================
__global__ __launch_bounds__(128)
void rownorm_k(float* __restrict__ X, int R, float numer) {
    __shared__ float s1[128];
    const int row = blockIdx.x;
    if (row >= R) return;
    float* x = X + (size_t)row * D_;
    float q = 0.f;
    for (int i = threadIdx.x; i < D_; i += 128) { float v = x[i]; q += v * v; }
    s1[threadIdx.x] = q; __syncthreads();
    for (int s = 64; s > 0; s >>= 1) {
        if (threadIdx.x < s) s1[threadIdx.x] += s1[threadIdx.x + s];
        __syncthreads();
    }
    const float f = numer * rsqrtf(s1[0]);
    for (int i = threadIdx.x; i < D_; i += 128) x[i] *= f;
}

// =====================================================================
// Attention: block = (seq, head). qkv f16 row-major [NSEQ*S, 3*512].
// f32 math internally; f16 output. Scores in dynamic LDS (S*S floats).
// =====================================================================
__global__ __launch_bounds__(256)
void attention_k(const _Float16* __restrict__ qkv, _Float16* __restrict__ out,
                 int S, int causal) {
    extern __shared__ float sc[];
    const int bh = blockIdx.x;
    const int b = bh / HEADS_, h = bh % HEADS_;
    const float scale = 0.125f;                    // 1/sqrt(64)
    const _Float16* base = qkv + (size_t)b * S * (3 * D_);
    const int tid = threadIdx.x, bs = blockDim.x;

    for (int i = tid; i < S * S; i += bs) {
        const int q = i / S, k = i % S;
        const _Float16* qp = base + (size_t)q * (3 * D_) + h * DH_;
        const _Float16* kp = base + (size_t)k * (3 * D_) + D_ + h * DH_;
        float acc = 0.f;
        #pragma unroll 8
        for (int d = 0; d < DH_; ++d) acc += (float)qp[d] * (float)kp[d];
        float sv = acc * scale;
        if (causal && (k > q)) sv += -1e9f;
        sc[i] = sv;
    }
    __syncthreads();
    for (int q = tid; q < S; q += bs) {
        float m = -INFINITY;
        for (int k = 0; k < S; ++k) m = fmaxf(m, sc[q * S + k]);
        float sum = 0.f;
        for (int k = 0; k < S; ++k) { float e = __expf(sc[q * S + k] - m); sc[q * S + k] = e; sum += e; }
        const float inv = 1.f / sum;
        for (int k = 0; k < S; ++k) sc[q * S + k] *= inv;
    }
    __syncthreads();
    for (int i = tid; i < S * DH_; i += bs) {
        const int q = i / DH_, d = i % DH_;
        float acc = 0.f;
        for (int k = 0; k < S; ++k)
            acc += sc[q * S + k] * (float)base[(size_t)k * (3 * D_) + 2 * D_ + h * DH_ + d];
        out[((size_t)(b * S + q)) * D_ + h * DH_ + d] = (_Float16)acc;
    }
}

// =====================================================================
// Entropy per row of logits [B, 1204]
// =====================================================================
__global__ __launch_bounds__(256)
void entropy_k(const float* __restrict__ logits, float* __restrict__ ent) {
    __shared__ float s1[256];
    const int row = blockIdx.x;
    const float* x = logits + (size_t)row * NCLS_;
    float acc = 0.f;
    for (int i = threadIdx.x; i < NCLS_; i += 256) {
        const float v = x[i];
        const float ls = (v < 0.f) ? (v - log1pf(__expf(v))) : (-log1pf(__expf(-v)));
        const float p  = 1.f / (1.f + __expf(-v));
        acc += -p * ls;
    }
    s1[threadIdx.x] = acc; __syncthreads();
    for (int s = 128; s > 0; s >>= 1) {
        if (threadIdx.x < s) s1[threadIdx.x] += s1[threadIdx.x + s];
        __syncthreads();
    }
    if (threadIdx.x == 0) ent[row] = s1[0];
}

// =====================================================================
// Top-32 over entropy[1000]. Single block; iterative argmax
// (ties -> smallest index, matching jax.lax.top_k ordering).
// =====================================================================
__global__ __launch_bounds__(256)
void topk_entropy_k(const float* __restrict__ ent, int* __restrict__ sel) {
    __shared__ float vals[1024];
    __shared__ float rv[256];
    __shared__ int   ri[256];
    const int tid = threadIdx.x;
    for (int i = tid; i < B_ROWS; i += 256) vals[i] = ent[i];
    __syncthreads();
    for (int it = 0; it < KSEL_; ++it) {
        float bv = -INFINITY; int bi = 0x7fffffff;
        for (int i = tid; i < B_ROWS; i += 256) {
            const float v = vals[i];
            if (v > bv || (v == bv && i < bi)) { bv = v; bi = i; }
        }
        rv[tid] = bv; ri[tid] = bi; __syncthreads();
        for (int s = 128; s > 0; s >>= 1) {
            if (tid < s) {
                if (rv[tid + s] > rv[tid] ||
                    (rv[tid + s] == rv[tid] && ri[tid + s] < ri[tid])) {
                    rv[tid] = rv[tid + s]; ri[tid] = ri[tid + s];
                }
            }
            __syncthreads();
        }
        if (tid == 0) { sel[it] = ri[0]; vals[ri[0]] = -INFINITY; }
        __syncthreads();
    }
}

// =====================================================================
// Top-5 classes per selected row (over first 1203 logits). 32 blocks.
// =====================================================================
__global__ __launch_bounds__(256)
void top5_classes_k(const float* __restrict__ logits, const int* __restrict__ sel,
                    int* __restrict__ topk) {
    __shared__ float vals[1216];
    __shared__ float rv[256];
    __shared__ int   ri[256];
    const int b = blockIdx.x, tid = threadIdx.x;
    const int row = sel[b];
    for (int i = tid; i < NC_; i += 256) vals[i] = logits[(size_t)row * NCLS_ + i];
    __syncthreads();
    for (int it = 0; it < KTOP_; ++it) {
        float bv = -INFINITY; int bi = 0x7fffffff;
        for (int i = tid; i < NC_; i += 256) {
            const float v = vals[i];
            if (v > bv || (v == bv && i < bi)) { bv = v; bi = i; }
        }
        rv[tid] = bv; ri[tid] = bi; __syncthreads();
        for (int s = 128; s > 0; s >>= 1) {
            if (tid < s) {
                if (rv[tid + s] > rv[tid] ||
                    (rv[tid + s] == rv[tid] && ri[tid + s] < ri[tid])) {
                    rv[tid] = rv[tid + s]; ri[tid] = ri[tid + s];
                }
            }
            __syncthreads();
        }
        if (tid == 0) { topk[b * KTOP_ + it] = ri[0]; vals[ri[0]] = -INFINITY; }
        __syncthreads();
    }
}

// =====================================================================
// Misc elementwise kernels
// =====================================================================
__global__ void gather_rows_f16_k(const float* __restrict__ src, const int* __restrict__ idx,
                                  _Float16* __restrict__ dst, int nrows) {
    const int i = blockIdx.x * blockDim.x + threadIdx.x;
    if (i >= nrows * D_) return;
    const int r = i / D_, d = i % D_;
    dst[i] = (_Float16)src[(size_t)idx[r] * D_ + d];
}

__global__ void build_agg_k(const float* __restrict__ ins_tok, const float* __restrict__ samp,
                            const float* __restrict__ exemplar, const int* __restrict__ topk,
                            float* __restrict__ aggx) {
    const int i = blockIdx.x * blockDim.x + threadIdx.x;
    const int total = NSEQ_ * (NINS_ + NEX_) * D_;
    if (i >= total) return;
    const int seq = i / ((NINS_ + NEX_) * D_);
    const int rem = i % ((NINS_ + NEX_) * D_);
    const int pos = rem / D_, d = rem % D_;
    const int s = seq / KTOP_, t = seq % KTOP_;
    float v;
    if (pos < NINS_) {
        v = ins_tok[pos * D_ + d] + samp[(size_t)s * D_ + d] * (1.f / TEMP_F);
    } else {
        const int cls = topk[s * KTOP_ + t];
        v = exemplar[((size_t)cls * NEX_ + (pos - NINS_)) * D_ + d];
    }
    aggx[i] = v;
}

__global__ void build_txt_k(const float* __restrict__ token_seq, const int* __restrict__ topk,
                            const float* __restrict__ aggx, const float* __restrict__ pos_emb,
                            float* __restrict__ X) {
    const int i = blockIdx.x * blockDim.x + threadIdx.x;
    const int total = NSEQ_ * LSEQ_ * D_;
    if (i >= total) return;
    const int seq = i / (LSEQ_ * D_);
    const int rem = i % (LSEQ_ * D_);
    const int pos = rem / D_, d = rem % D_;
    const int s = seq / KTOP_, t = seq % KTOP_;
    const int cls = topk[s * KTOP_ + t];
    float v;
    if (pos < 2)        v = token_seq[((size_t)cls * LSEQ_ + pos) * D_ + d];
    else if (pos < 6)   v = aggx[((size_t)(seq * (NINS_ + NEX_) + (pos - 2))) * D_ + d];
    else                v = token_seq[((size_t)cls * LSEQ_ + (pos - NINS_)) * D_ + d];
    X[i] = v + pos_emb[pos * D_ + d];
}

__global__ void gather_eos_f16_k(const _Float16* __restrict__ lnh, const int* __restrict__ eos_index,
                                 const int* __restrict__ topk, _Float16* __restrict__ dst) {
    const int i = blockIdx.x * blockDim.x + threadIdx.x;
    if (i >= NSEQ_ * D_) return;
    const int seq = i / D_, d = i % D_;
    const int s = seq / KTOP_, t = seq % KTOP_;
    const int e = eos_index[topk[s * KTOP_ + t]] + 2 * NINS_;
    dst[i] = lnh[((size_t)(seq * LSEQ_ + e)) * D_ + d];
}

__global__ __launch_bounds__(128)
void rerank_k(const float* __restrict__ samp, const float* __restrict__ feats,
              float* __restrict__ out_rerank) {
    __shared__ float s1[128];
    const int seq = blockIdx.x;
    const int s = seq / KTOP_;
    float acc = 0.f;
    for (int d = threadIdx.x; d < D_; d += 128)
        acc += samp[(size_t)s * D_ + d] * feats[(size_t)seq * D_ + d];
    s1[threadIdx.x] = acc; __syncthreads();
    for (int st = 64; st > 0; st >>= 1) {
        if (threadIdx.x < st) s1[threadIdx.x] += s1[threadIdx.x + st];
        __syncthreads();
    }
    if (threadIdx.x == 0) out_rerank[seq] = s1[0];
}

__global__ void copy_k(const float* __restrict__ src, float* __restrict__ dst, int n) {
    const int i = blockIdx.x * blockDim.x + threadIdx.x;
    if (i < n) dst[i] = src[i];
}

__global__ void write_sel_k(const int* __restrict__ sel, float* __restrict__ out) {
    const int i = threadIdx.x;
    if (i < KSEL_) out[i] = (float)sel[i];
}

__global__ void fuse_scatter_k(const float* __restrict__ logits, const float* __restrict__ rerank,
                               const int* __restrict__ sel, const int* __restrict__ topk,
                               float* __restrict__ out_final) {
    const int i = threadIdx.x;
    if (i >= KSEL_ * KTOP_) return;
    const int s = i / KTOP_, t = i % KTOP_;
    const int row = sel[s];
    const int cls = topk[s * KTOP_ + t];
    const float sv = logits[(size_t)row * NCLS_ + cls];
    const float rr = rerank[i];
    const float sig_s = 1.f / (1.f + __expf(-sv));
    const float sig_r = 1.f / (1.f + __expf(-rr));
    float fused = powf(sig_s, 0.75f) * powf(sig_r, 0.25f);
    fused = fminf(fmaxf(fused, 0.f), 1.f);
    const float x1 = fmaxf(fused, 1e-5f);
    const float x2 = fmaxf(1.f - fused, 1e-5f);
    out_final[(size_t)row * NCLS_ + cls] = logf(x1 / x2);
}

// =====================================================================
// Host-side helpers
// =====================================================================
enum { F_BIAS = 1, F_RESID = 2, F_GELU = 4, F_OUT16 = 8 };

static void gemm(hipStream_t st, const _Float16* A, const _Float16* WT, const float* bias,
                 const float* resid, void* C, int M, int N, int K, int flags) {
    const int strips = ((M + 15) / 16) * ((N + 63) / 64);
    const int blocks = (strips + 7) / 8;
    switch (flags) {
        case 0:
            gemm_wmma_k<false, false, false, false><<<blocks, 256, 0, st>>>(A, WT, bias, resid, C, M, N, K); break;
        case F_BIAS:
            gemm_wmma_k<true, false, false, false><<<blocks, 256, 0, st>>>(A, WT, bias, resid, C, M, N, K); break;
        case F_BIAS | F_OUT16:
            gemm_wmma_k<true, false, false, true><<<blocks, 256, 0, st>>>(A, WT, bias, resid, C, M, N, K); break;
        case F_BIAS | F_RESID:
            gemm_wmma_k<true, true, false, false><<<blocks, 256, 0, st>>>(A, WT, bias, resid, C, M, N, K); break;
        case F_BIAS | F_GELU | F_OUT16:
            gemm_wmma_k<true, false, true, true><<<blocks, 256, 0, st>>>(A, WT, bias, resid, C, M, N, K); break;
        default: break;
    }
}

static void transpose_cvt(hipStream_t st, const float* W, _Float16* WT, int K, int N) {
    dim3 g((K + 31) / 32, (N + 31) / 32);
    transpose_cvt_k<<<g, 256, 0, st>>>(W, WT, K, N);
}

struct TfW { const _Float16 *qkvT, *outT, *fcT, *prT; };

// Pre-LN residual transformer, NL_ layers.
// P[12]: ln1_g, ln1_b, qkv_w, qkv_b, out_w, out_b, ln2_g, ln2_b, fc_w, fc_b, pr_w, pr_b
static void run_transformer(hipStream_t st, float* Xb, _Float16* LNH, _Float16* BUFH,
                            int nseq, int S, const float* const* P, const TfW& W, int causal) {
    const int M = nseq * S;
    for (int l = 0; l < NL_; ++l) {
        layernorm_k<<<M, 128, 0, st>>>(Xb, P[0] + (size_t)l * D_, P[1] + (size_t)l * D_, LNH, M);
        gemm(st, LNH, W.qkvT + (size_t)l * 3 * D_ * D_, P[3] + (size_t)l * 3 * D_,
             nullptr, BUFH, M, 3 * D_, D_, F_BIAS | F_OUT16);
        attention_k<<<nseq * HEADS_, 256, (size_t)S * S * sizeof(float), st>>>(BUFH, LNH, S, causal);
        gemm(st, LNH, W.outT + (size_t)l * D_ * D_, P[5] + (size_t)l * D_,
             Xb, Xb, M, D_, D_, F_BIAS | F_RESID);
        layernorm_k<<<M, 128, 0, st>>>(Xb, P[6] + (size_t)l * D_, P[7] + (size_t)l * D_, LNH, M);
        gemm(st, LNH, W.fcT + (size_t)l * MLP_ * D_, P[9] + (size_t)l * MLP_,
             nullptr, BUFH, M, MLP_, D_, F_BIAS | F_GELU | F_OUT16);
        gemm(st, BUFH, W.prT + (size_t)l * D_ * MLP_, P[11] + (size_t)l * D_,
             Xb, Xb, M, D_, MLP_, F_BIAS | F_RESID);
    }
}

extern "C" void kernel_launch(void* const* d_in, const int* in_sizes, int n_in,
                              void* d_out, int out_size, void* d_ws, size_t ws_size,
                              hipStream_t stream) {
    (void)in_sizes; (void)n_in; (void)out_size; (void)ws_size;

    // ---- inputs (setup_inputs dict order) ----
    const float* x         = (const float*)d_in[0];
    const float* linear_w  = (const float*)d_in[1];
    const float* linear_b  = (const float*)d_in[2];
    const float* zs_w      = (const float*)d_in[3];
    const float* sample_w  = (const float*)d_in[4];
    const float* sample_b  = (const float*)d_in[5];
    const float* ins_tok   = (const float*)d_in[6];
    const float* token_seq = (const float*)d_in[7];
    const float* exemplar  = (const float*)d_in[8];
    const int*   eos_index = (const int*)d_in[9];
    const float* aggP[12]; for (int i = 0; i < 12; ++i) aggP[i] = (const float*)d_in[10 + i];
    const float* pos_emb   = (const float*)d_in[22];
    const float* txtP[12]; for (int i = 0; i < 12; ++i) txtP[i] = (const float*)d_in[23 + i];
    const float* lnf_g     = (const float*)d_in[35];
    const float* lnf_b     = (const float*)d_in[36];
    const float* text_proj = (const float*)d_in[37];

    // ---- workspace carve ----
    // f32 region
    float* ws   = (float*)d_ws;
    float* Xb   = ws;                                    // 12320*512
    float* H    = Xb   + (size_t)NSEQ_ * LSEQ_ * D_;     // 1000*512
    float* AGGX = H    + (size_t)B_ROWS * D_;            // 1440*512
    float* SAMP = AGGX + (size_t)NSEQ_ * (NINS_ + NEX_) * D_;  // 32*512
    float* FEAT = SAMP + (size_t)KSEL_ * D_;             // 160*512
    float* ENT  = FEAT + (size_t)NSEQ_ * D_;             // 1024
    int*   SEL  = (int*)(ENT + 1024);                    // 32
    int*   TOPK = SEL + KSEL_;                           // 160
    // f16 region (16B-aligned: preceding float+int counts are multiples of 4)
    _Float16* LNH   = (_Float16*)(TOPK + NSEQ_);         // 12320*512
    _Float16* BUFH  = LNH   + (size_t)NSEQ_ * LSEQ_ * D_;       // 12320*2048 (qkv/mlp shared)
    _Float16* XH    = BUFH  + (size_t)NSEQ_ * LSEQ_ * MLP_;     // 1000*1024
    _Float16* HH    = XH    + (size_t)B_ROWS * DIN_;            // 1000*512
    _Float16* HSELH = HH    + (size_t)B_ROWS * D_;              // 32*512
    _Float16* EOSRH = HSELH + (size_t)KSEL_ * D_;               // 160*512
    _Float16* WlinT = EOSRH + (size_t)NSEQ_ * D_;               // 512*1024
    _Float16* WzsT  = WlinT + (size_t)D_ * DIN_;                // 1204*512
    _Float16* WsmpT = WzsT  + (size_t)NCLS_ * D_;               // 512*512
    _Float16* WprjT = WsmpT + (size_t)D_ * D_;                  // 512*512
    _Float16* AQKVT = WprjT + (size_t)D_ * D_;                  // 4*1536*512
    _Float16* AOUTT = AQKVT + (size_t)NL_ * 3 * D_ * D_;        // 4*512*512
    _Float16* AFCT  = AOUTT + (size_t)NL_ * D_ * D_;            // 4*2048*512
    _Float16* APRT  = AFCT  + (size_t)NL_ * MLP_ * D_;          // 4*512*2048
    _Float16* TQKVT = APRT  + (size_t)NL_ * D_ * MLP_;
    _Float16* TOUTT = TQKVT + (size_t)NL_ * 3 * D_ * D_;
    _Float16* TFCT  = TOUTT + (size_t)NL_ * D_ * D_;
    _Float16* TPRT  = TFCT  + (size_t)NL_ * MLP_ * D_;

    // ---- output carve (tuple order, flat) ----
    float* out        = (float*)d_out;
    float* out_logits = out;                                    // 1000*1204
    float* out_rerank = out_logits + (size_t)B_ROWS * NCLS_;    // 160
    float* out_sel    = out_rerank + NSEQ_;                     // 32
    float* out_final  = out_sel + KSEL_;                        // 1000*1204

    // 0) weight pre-pass: f32 [K,N] -> f16 [N,K]
    transpose_cvt(stream, linear_w,  WlinT, DIN_, D_);
    transpose_cvt(stream, zs_w,      WzsT,  D_, NCLS_);
    transpose_cvt(stream, sample_w,  WsmpT, D_, D_);
    transpose_cvt(stream, text_proj, WprjT, D_, D_);
    for (int l = 0; l < NL_; ++l) {
        transpose_cvt(stream, aggP[2]  + (size_t)l * D_ * 3 * D_, AQKVT + (size_t)l * 3 * D_ * D_, D_, 3 * D_);
        transpose_cvt(stream, aggP[4]  + (size_t)l * D_ * D_,     AOUTT + (size_t)l * D_ * D_,     D_, D_);
        transpose_cvt(stream, aggP[8]  + (size_t)l * D_ * MLP_,   AFCT  + (size_t)l * MLP_ * D_,   D_, MLP_);
        transpose_cvt(stream, aggP[10] + (size_t)l * MLP_ * D_,   APRT  + (size_t)l * D_ * MLP_,   MLP_, D_);
        transpose_cvt(stream, txtP[2]  + (size_t)l * D_ * 3 * D_, TQKVT + (size_t)l * 3 * D_ * D_, D_, 3 * D_);
        transpose_cvt(stream, txtP[4]  + (size_t)l * D_ * D_,     TOUTT + (size_t)l * D_ * D_,     D_, D_);
        transpose_cvt(stream, txtP[8]  + (size_t)l * D_ * MLP_,   TFCT  + (size_t)l * MLP_ * D_,   D_, MLP_);
        transpose_cvt(stream, txtP[10] + (size_t)l * MLP_ * D_,   TPRT  + (size_t)l * D_ * MLP_,   MLP_, D_);
    }
    const TfW aggW = { AQKVT, AOUTT, AFCT, APRT };
    const TfW txtW = { TQKVT, TOUTT, TFCT, TPRT };

    // 1) base classifier: h = x @ linear_w + b ; h = TEMP*h/||h||
    {
        const int n = B_ROWS * DIN_;
        cvt_f16_k<<<(n + 255) / 256, 256, 0, stream>>>(x, XH, n);
    }
    gemm(stream, XH, WlinT, linear_b, nullptr, H, B_ROWS, D_, DIN_, F_BIAS);
    rownorm_k<<<B_ROWS, 128, 0, stream>>>(H, B_ROWS, TEMP_F);

    // 2) logits = h @ zs_weight  (written directly to output segment 0)
    {
        const int n = B_ROWS * D_;
        cvt_f16_k<<<(n + 255) / 256, 256, 0, stream>>>(H, HH, n);
    }
    gemm(stream, HH, WzsT, nullptr, nullptr, out_logits, B_ROWS, NCLS_, D_, 0);

    // 3) entropy + top-32 selection
    entropy_k<<<B_ROWS, 256, 0, stream>>>(out_logits, ENT);
    topk_entropy_k<<<1, 256, 0, stream>>>(ENT, SEL);

    // 4) top-5 classes per selected proposal
    top5_classes_k<<<KSEL_, 256, 0, stream>>>(out_logits, SEL, TOPK);

    // 5) sampled_x = h[select] @ sample_w + sample_b
    gather_rows_f16_k<<<(KSEL_ * D_ + 255) / 256, 256, 0, stream>>>(H, SEL, HSELH, KSEL_);
    gemm(stream, HSELH, WsmpT, sample_b, nullptr, SAMP, KSEL_, D_, D_, F_BIAS);

    // 6) aggregator input + transformer (160 seqs x 9 tokens)
    {
        const int total = NSEQ_ * (NINS_ + NEX_) * D_;
        build_agg_k<<<(total + 255) / 256, 256, 0, stream>>>(ins_tok, SAMP, exemplar, TOPK, AGGX);
    }
    run_transformer(stream, AGGX, LNH, BUFH, NSEQ_, NINS_ + NEX_, aggP, aggW, /*causal=*/0);

    // 7) splice instance tokens into prompts + pos_emb; text transformer (causal)
    {
        const int total = NSEQ_ * LSEQ_ * D_;
        build_txt_k<<<(total + 255) / 256, 256, 0, stream>>>(token_seq, TOPK, AGGX, pos_emb, Xb);
    }
    run_transformer(stream, Xb, LNH, BUFH, NSEQ_, LSEQ_, txtP, txtW, /*causal=*/1);

    // 8) final LN, eos gather, projection, normalize
    layernorm_k<<<NSEQ_ * LSEQ_, 128, 0, stream>>>(Xb, lnf_g, lnf_b, LNH, NSEQ_ * LSEQ_);
    gather_eos_f16_k<<<(NSEQ_ * D_ + 255) / 256, 256, 0, stream>>>(LNH, eos_index, TOPK, EOSRH);
    gemm(stream, EOSRH, WprjT, nullptr, nullptr, FEAT, NSEQ_, D_, D_, 0);
    rownorm_k<<<NSEQ_, 128, 0, stream>>>(FEAT, NSEQ_, 1.0f);

    // 9) rerank logits
    rerank_k<<<NSEQ_, 128, 0, stream>>>(SAMP, FEAT, out_rerank);

    // 10) select_index output (as float), logits_final = copy + fused scatter
    write_sel_k<<<1, 32, 0, stream>>>(SEL, out_sel);
    {
        const int n = B_ROWS * NCLS_;
        copy_k<<<(n + 255) / 256, 256, 0, stream>>>(out_logits, out_final, n);
    }
    fuse_scatter_k<<<1, 256, 0, stream>>>(out_logits, out_rerank, SEL, TOPK, out_final);
}
